// VocabGNN_12876311953626
// MI455X (gfx1250) — matrix operand order repaired and checked by reading, not compiled, not run on hardware
//
#include <hip/hip_runtime.h>
#include <hip/hip_bf16.h>
#include <stdint.h>

typedef __attribute__((ext_vector_type(16))) __bf16 v16bf;
typedef __attribute__((ext_vector_type(8)))  __bf16 v8bf;
typedef __attribute__((ext_vector_type(8)))  float  v8f;

#define HID 256
#define NH  4
#define HD  64

__device__ __forceinline__ unsigned short f32_to_bf16_rne(float f) {
    unsigned int u = __float_as_uint(f);
    unsigned int lsb = (u >> 16) & 1u;
    u += 0x7fffu + lsb;              // round-to-nearest-even
    return (unsigned short)(u >> 16);
}

__global__ void zero_f32(float* __restrict__ p, long long n) {
    long long i = (long long)blockIdx.x * blockDim.x + threadIdx.x;
    if (i < n) p[i] = 0.f;
}

__global__ void cvt_f32_bf16(const float* __restrict__ in, unsigned short* __restrict__ out,
                             long long n) {
    long long i = (long long)blockIdx.x * blockDim.x + threadIdx.x;
    if (i < n) out[i] = f32_to_bf16_rne(in[i]);
}

// in: [R x C] f32 row-major -> out: [C x R] bf16 row-major (transpose + convert)
__global__ void cvt_f32_bf16_T(const float* __restrict__ in, unsigned short* __restrict__ out,
                               int R, int C) {
    long long i = (long long)blockIdx.x * blockDim.x + threadIdx.x;
    if (i >= (long long)R * C) return;
    int r = (int)(i / C), c = (int)(i % C);
    out[(size_t)c * R + r] = f32_to_bf16_rne(in[i]);
}

// H[src[e], c] += val[e] * W[dst[e], c]   (one block per edge, c = threadIdx.x)
__global__ void gat_scatter_H(const int* __restrict__ src, const int* __restrict__ dst,
                              const float* __restrict__ val, const float* __restrict__ W,
                              float* __restrict__ H) {
    int e = blockIdx.x, c = threadIdx.x;
    int s = src[e], d = dst[e];
    float v = val[e];
    atomicAdd(&H[(size_t)s * HID + c], v * W[(size_t)d * HID + c]);
}

// per (edge, head): e = leaky_relu(dot(Hs,a1)+dot(Hd,a2)); ex=exp(e); denom[dst,h]+=ex
__global__ void gat_edge_attn(const int* __restrict__ src, const int* __restrict__ dst,
                              const float* __restrict__ H, const float* __restrict__ a,
                              float* __restrict__ exb, float* __restrict__ denom, int E) {
    long long idx = (long long)blockIdx.x * blockDim.x + threadIdx.x;
    if (idx >= (long long)E * NH) return;
    int e = (int)(idx >> 2), h = (int)(idx & 3);
    const float* Hs = H + (size_t)src[e] * HID + h * HD;
    const float* Hd = H + (size_t)dst[e] * HID + h * HD;
    const float* as = a + h * 2 * HD;
    float acc = 0.f;
    #pragma unroll 8
    for (int i = 0; i < HD; ++i)
        acc += Hs[i] * as[i] + Hd[i] * as[HD + i];
    float lr = acc >= 0.f ? acc : 0.2f * acc;
    float ex = expf(lr);
    exb[idx] = ex;
    atomicAdd(&denom[(size_t)dst[e] * NH + h], ex);
}

// HS[dst[e], c] += (ex[e,h]/(denom[dst,h]+1e-9)) * H[src[e], c]
__global__ void gat_agg(const int* __restrict__ src, const int* __restrict__ dst,
                        const float* __restrict__ H, const float* __restrict__ exb,
                        const float* __restrict__ denom, float* __restrict__ HS) {
    int e = blockIdx.x, c = threadIdx.x;
    int h = c >> 6;
    int d = dst[e];
    float alpha = exb[(size_t)e * NH + h] / (denom[(size_t)d * NH + h] + 1e-9f);
    atomicAdd(&HS[(size_t)d * HID + c], alpha * H[(size_t)src[e] * HID + c]);
}

// D[M x N](f32) = A[M x K](bf16 row-major) x Bt[N x K](bf16 row-major)^T (+ bias)
// One wave per (16 x 16*NT) output strip; K % 32 == 0; all lanes K-contiguous.
template <int NT>
__global__ void wmma_gemm_bt(const unsigned short* __restrict__ A,
                             const unsigned short* __restrict__ Bt,
                             const float* __restrict__ bias,
                             float* __restrict__ D, int N, int K) {
    int mBase  = blockIdx.x * 16;
    int nBase0 = blockIdx.y * 16 * NT;
    int lane = threadIdx.x;
    int r  = lane & 15;
    int hi = lane >> 4;                 // 0: lanes 0-15, 1: lanes 16-31
    const __bf16* Ap = reinterpret_cast<const __bf16*>(A);
    const __bf16* Bp = reinterpret_cast<const __bf16*>(Bt);
    const __bf16* arow = Ap + (size_t)(mBase + r) * K;
    const __bf16* brow[NT];
    #pragma unroll
    for (int t = 0; t < NT; ++t)
        brow[t] = Bp + (size_t)(nBase0 + t * 16 + r) * K;

    v8f c[NT];
    #pragma unroll
    for (int t = 0; t < NT; ++t) c[t] = (v8f){};

    for (int k0 = 0; k0 < K; k0 += 32) {
        // A 16x32 fragment: lanes<16 K 0..15; lanes>=16 K 16..31
        int kA = k0 + hi * 16;
        v8bf alo = *(const v8bf*)(arow + kA);
        v8bf ahi = *(const v8bf*)(arow + kA + 8);
        v16bf av = __builtin_shufflevector(alo, ahi, 0, 1, 2, 3, 4, 5, 6, 7,
                                           8, 9, 10, 11, 12, 13, 14, 15);
        // B 32x16 fragment: lane n holds column n; lanes<16 K 0..15, lanes>=16 K 16..31
        // Bt row (n) is K-contiguous -> single 32-byte fragment load
        int kB = k0 + hi * 16;
        #pragma unroll
        for (int t = 0; t < NT; ++t) {
            v16bf bv = *(const v16bf*)(brow[t] + kB);
            c[t] = __builtin_amdgcn_wmma_f32_16x16x32_bf16(false, av, false, bv,
                                                           (short)0, c[t], false, false);
        }
    }
    // C/D layout: lanes<16 rows M 0..7, lanes>=16 rows M 8..15, N = lane&15
    #pragma unroll
    for (int t = 0; t < NT; ++t) {
        int nb = nBase0 + t * 16 + r;
        float bb = bias ? bias[nb] : 0.f;
        #pragma unroll
        for (int i = 0; i < 8; ++i)
            D[(size_t)(mBase + 8 * hi + i) * N + nb] = c[t][i] + bb;
    }
}

extern "C" void kernel_launch(void* const* d_in, const int* in_sizes, int n_in,
                              void* d_out, int out_size, void* d_ws, size_t ws_size,
                              hipStream_t stream) {
    const float* X    = (const float*)d_in[0];
    const float* W0   = (const float*)d_in[1];
    const float* a0   = (const float*)d_in[2];
    const float* W1   = (const float*)d_in[3];
    const float* a1   = (const float*)d_in[4];
    const float* fcw  = (const float*)d_in[5];
    const float* fcb  = (const float*)d_in[6];
    const float* val0 = (const float*)d_in[7];
    const float* val1 = (const float*)d_in[8];
    const int*   src0 = (const int*)d_in[9];
    const int*   dst0 = (const int*)d_in[10];
    const int*   src1 = (const int*)d_in[11];
    const int*   dst1 = (const int*)d_in[12];

    const int V    = in_sizes[1] / HID;   // 20000 (= 625*32, exact K tiling)
    const int E    = in_sizes[7];         // 640000
    const int M    = in_sizes[0] / V;     // 1024
    const int OUTD = in_sizes[6];         // 768

    char* ws = (char*)d_ws; (void)ws_size; (void)n_in; (void)out_size;
    size_t off = 0;
    auto take = [&](size_t bytes) {
        char* p = ws + off;
        off = (off + bytes + 255) & ~(size_t)255;
        return p;
    };
    float*          HS      = (float*)take((size_t)V * HID * 4);
    float*          Htmp    = (float*)take((size_t)V * HID * 4);
    float*          denom   = (float*)take((size_t)V * NH * 4);
    float*          exb     = (float*)take((size_t)E * NH * 4);
    unsigned short* Xbf     = (unsigned short*)take((size_t)M * V * 2);
    unsigned short* HST     = (unsigned short*)take((size_t)V * HID * 2);  // [HID x V]
    float*          fused   = (float*)take((size_t)M * HID * 4);
    unsigned short* fusedbf = (unsigned short*)take((size_t)M * HID * 2);
    unsigned short* fcwbf   = (unsigned short*)take((size_t)OUTD * HID * 2);

    long long nHS = (long long)V * HID;
    zero_f32<<<(int)((nHS + 255) / 256), 256, 0, stream>>>(HS, nHS);

    for (int l = 0; l < 2; ++l) {
        const int*   src = l ? src1 : src0;
        const int*   dst = l ? dst1 : dst0;
        const float* val = l ? val1 : val0;
        const float* W   = l ? W1   : W0;
        const float* a   = l ? a1   : a0;

        zero_f32<<<(int)((nHS + 255) / 256), 256, 0, stream>>>(Htmp, nHS);
        long long nD = (long long)V * NH;
        zero_f32<<<(int)((nD + 255) / 256), 256, 0, stream>>>(denom, nD);

        gat_scatter_H<<<E, HID, 0, stream>>>(src, dst, val, W, Htmp);
        long long nEH = (long long)E * NH;
        gat_edge_attn<<<(int)((nEH + 255) / 256), 256, 0, stream>>>(src, dst, Htmp, a,
                                                                    exb, denom, E);
        gat_agg<<<E, HID, 0, stream>>>(src, dst, Htmp, exb, denom, HS);
    }

    long long nX = (long long)M * V;
    cvt_f32_bf16<<<(int)((nX + 255) / 256), 256, 0, stream>>>(X, Xbf, nX);
    // HS [V x HID] -> HST [HID x V] bf16 (K-contiguous B operand)
    cvt_f32_bf16_T<<<(int)((nHS + 255) / 256), 256, 0, stream>>>(HS, HST, V, HID);

    dim3 g1(M / 16, HID / (16 * 4));
    wmma_gemm_bt<4><<<g1, 32, 0, stream>>>(Xbf, HST, nullptr, fused, HID, V);

    long long nF = (long long)M * HID;
    cvt_f32_bf16<<<(int)((nF + 255) / 256), 256, 0, stream>>>(fused, fusedbf, nF);
    long long nW = (long long)OUTD * HID;
    cvt_f32_bf16<<<(int)((nW + 255) / 256), 256, 0, stream>>>(fcw, fcwbf, nW);

    dim3 g2(M / 16, OUTD / (16 * 4));
    wmma_gemm_bt<4><<<g2, 32, 0, stream>>>(fusedbf, fcwbf, fcb, (float*)d_out, OUTD, HID);
}
